// MoELayer_27754078667461
// MI455X (gfx1250) — matrix operand order repaired and checked by reading, not compiled, not run on hardware
//
#include <hip/hip_runtime.h>
#include <hip/hip_bf16.h>
#include <cmath>

// ---------------------------------------------------------------------------
// MoE (top-2 of 8 experts), D=1024, F=4096, T=8192 tokens.
// gate -> plan -> scatter -> (W -> bf16 transposed) ->
// fused WMMA expert FFN (GEMM1+GELU+GEMM2 in 320KB LDS) -> combine.
// Round 3: A fragments rotate through register rings loaded 2 steps ahead and
// placed after the consuming WMMA, eliminating the WMMA->ds_load WAR v_nops
// and the s_wait_dscnt 0x0 before every WMMA.
// ---------------------------------------------------------------------------

#define T_TOK   8192
#define D_MODEL 1024
#define D_FF    4096
#define N_EXP   8
#define TILE_M  64          // tokens per expert tile
#define FCHUNK  256         // F columns processed per LDS round
#define MAX_TILES 264       // sum ceil(n_e/64) <= 16384/64 + 8
#define AXS     1032        // LDS row stride (bf16 elems) for x tile (pad 8)
#define ACS     264         // LDS row stride for activation chunk (pad 8)
#define SMEM_BYTES (TILE_M*AXS*2 + TILE_M*ACS*2 + TILE_M*12)

typedef __attribute__((ext_vector_type(16))) __bf16 v16bf;
typedef __attribute__((ext_vector_type(8)))  float  v8f;

union Frag { v16bf v; uint4 q[2]; };

__device__ __forceinline__ v8f zerov8() {
  v8f z;
#pragma unroll
  for (int i = 0; i < 8; ++i) z[i] = 0.0f;
  return z;
}

// A-matrix 16x32 bf16 fragment (matches B order): lane holds row M=lane&15;
// 16 consecutive K starting at k0 + 2*hi8 (hi8=(lane>=16)*8)
__device__ __forceinline__ v16bf ldfragA(const __bf16* rowp, int k0, int hi8) {
  Frag f;
  f.q[0] = *reinterpret_cast<const uint4*>(rowp + k0 + 2 * hi8);
  f.q[1] = *reinterpret_cast<const uint4*>(rowp + k0 + 8 + 2 * hi8);
  return f.v;
}

// B-matrix 32x16 bf16 fragment (contiguous k order): lane holds col N=lane&15;
// 16 consecutive K starting at k0 + (lane>=16)*16 -> one 32B contiguous region.
__device__ __forceinline__ v16bf ldfragB(const __bf16* colp, int k0, int hi16) {
  Frag f;
  const uint4* p = reinterpret_cast<const uint4*>(colp + k0 + hi16);
  f.q[0] = p[0];
  f.q[1] = p[1];
  return f.v;
}

#define WMMA_BF16(A, B, C) \
  __builtin_amdgcn_wmma_f32_16x16x32_bf16(false, (A), false, (B), (short)0, (C), false, false)

// ---------------------------------------------------------------------------
// K1: gating. One wave per token: logits, softmax, top-2, counts.
// ---------------------------------------------------------------------------
__global__ __launch_bounds__(256) void gate_kernel(
    const float* __restrict__ x, const float* __restrict__ Wg, const float* __restrict__ bg,
    float* __restrict__ gate_out, int* __restrict__ counts,
    int* __restrict__ tokE, float* __restrict__ tokW)
{
  int lane = threadIdx.x & 31;
  int wv   = threadIdx.x >> 5;
  int t    = blockIdx.x * 8 + wv;

  float acc[N_EXP];
#pragma unroll
  for (int e = 0; e < N_EXP; ++e) acc[e] = 0.0f;

  const float* xr = x + (size_t)t * D_MODEL;
  for (int k = lane; k < D_MODEL; k += 32) {
    float xv = xr[k];
    const float4* wr = reinterpret_cast<const float4*>(Wg + (size_t)k * N_EXP);
    float4 a = wr[0], b = wr[1];
    acc[0] += xv * a.x; acc[1] += xv * a.y; acc[2] += xv * a.z; acc[3] += xv * a.w;
    acc[4] += xv * b.x; acc[5] += xv * b.y; acc[6] += xv * b.z; acc[7] += xv * b.w;
  }
#pragma unroll
  for (int off = 16; off > 0; off >>= 1) {
#pragma unroll
    for (int e = 0; e < N_EXP; ++e) acc[e] += __shfl_xor(acc[e], off, 32);
  }

  if (lane == 0) {
    float l[N_EXP];
#pragma unroll
    for (int e = 0; e < N_EXP; ++e) { l[e] = acc[e] + bg[e]; gate_out[(size_t)t * N_EXP + e] = l[e]; }
    float mx = l[0];
#pragma unroll
    for (int e = 1; e < N_EXP; ++e) mx = fmaxf(mx, l[e]);
    float p[N_EXP], s = 0.0f;
#pragma unroll
    for (int e = 0; e < N_EXP; ++e) { p[e] = __expf(l[e] - mx); s += p[e]; }
    // top-2 over probs == top-2 over logits; lowest index wins ties (strict >)
    int e0 = 0; float b0 = p[0];
#pragma unroll
    for (int e = 1; e < N_EXP; ++e) if (p[e] > b0) { b0 = p[e]; e0 = e; }
    int e1 = -1; float b1v = -1.0f;
#pragma unroll
    for (int e = 0; e < N_EXP; ++e) if (e != e0 && p[e] > b1v) { b1v = p[e]; e1 = e; }
    float inv = 1.0f / s;
    tokE[t * 2 + 0] = e0;  tokW[t * 2 + 0] = b0 * inv;
    tokE[t * 2 + 1] = e1;  tokW[t * 2 + 1] = b1v * inv;
    atomicAdd(&counts[e0], 1);
    atomicAdd(&counts[e1], 1);
  }
}

// ---------------------------------------------------------------------------
// K2: plan. Prefix sums + tile descriptors (single thread; trivial sizes).
// ---------------------------------------------------------------------------
__global__ void plan_kernel(const int* __restrict__ counts, int* __restrict__ offsets,
                            int* __restrict__ tileE, int* __restrict__ tileS,
                            int* __restrict__ tileC, int* __restrict__ numTiles)
{
  if (threadIdx.x == 0 && blockIdx.x == 0) {
    int off = 0, nt = 0;
    for (int e = 0; e < N_EXP; ++e) {
      offsets[e] = off;
      int n = counts[e];
      for (int s = 0; s < n; s += TILE_M) {
        tileE[nt] = e;
        tileS[nt] = off + s;
        tileC[nt] = (n - s < TILE_M) ? (n - s) : TILE_M;
        ++nt;
      }
      off += n;
    }
    offsets[N_EXP] = off;
    *numTiles = nt;
  }
}

// ---------------------------------------------------------------------------
// K3: scatter tokens into expert buckets (order nondeterministic; math isn't).
// ---------------------------------------------------------------------------
__global__ __launch_bounds__(256) void scatter_kernel(
    const int* __restrict__ tokE, const float* __restrict__ tokW,
    const int* __restrict__ offsets, int* __restrict__ fill,
    int* __restrict__ rowTok, int* __restrict__ rowSlot, float* __restrict__ rowW)
{
  int t = blockIdx.x * 256 + threadIdx.x;
#pragma unroll
  for (int k = 0; k < 2; ++k) {
    int e = tokE[t * 2 + k];
    int pos = offsets[e] + atomicAdd(&fill[e], 1);
    rowTok[pos]  = t;
    rowSlot[pos] = k;
    rowW[pos]    = tokW[t * 2 + k];
  }
}

// ---------------------------------------------------------------------------
// K4: convert+transpose f32 [E][R][C] -> bf16 [E][C][R] (B columns contiguous).
// ---------------------------------------------------------------------------
__global__ __launch_bounds__(256) void transpose_cvt_kernel(
    const float* __restrict__ src, __bf16* __restrict__ dst, int R, int C)
{
  __shared__ float tile[32][33];
  int e = blockIdx.z;
  int c0 = blockIdx.x * 32, r0 = blockIdx.y * 32;
  const float* s = src + (size_t)e * R * C;
  __bf16* d = dst + (size_t)e * R * C;
#pragma unroll
  for (int i = threadIdx.y; i < 32; i += 8)
    tile[i][threadIdx.x] = s[(size_t)(r0 + i) * C + c0 + threadIdx.x];
  __syncthreads();
#pragma unroll
  for (int i = threadIdx.y; i < 32; i += 8)
    d[(size_t)(c0 + i) * R + r0 + threadIdx.x] = (__bf16)tile[threadIdx.x][i];
}

// ---------------------------------------------------------------------------
// K5: fused expert FFN. 512 threads = 16 waves, tile = 64 tokens.
// ---------------------------------------------------------------------------
__global__ __launch_bounds__(512) void moe_expert_kernel(
    const float* __restrict__ x,
    const __bf16* __restrict__ W1t,   // [E][F][D] bf16
    const __bf16* __restrict__ W2t,   // [E][D][F] bf16
    const float* __restrict__ b1, const float* __restrict__ b2,
    const int* __restrict__ numTiles,
    const int* __restrict__ tileE, const int* __restrict__ tileS, const int* __restrict__ tileC,
    const int* __restrict__ rowTok, const int* __restrict__ rowSlot, const float* __restrict__ rowW,
    float* __restrict__ ybuf)
{
  int bid = blockIdx.x;
  if (bid >= *numTiles) return;
  int e  = tileE[bid];
  int ts = tileS[bid];
  int tc = tileC[bid];

  extern __shared__ char smem[];
  __bf16* Ax   = reinterpret_cast<__bf16*>(smem);                       // [64][AXS]
  __bf16* Ac   = reinterpret_cast<__bf16*>(smem + TILE_M * AXS * 2);    // [64][ACS]
  int*    tTok = reinterpret_cast<int*>(smem + TILE_M * AXS * 2 + TILE_M * ACS * 2);
  int*    tSlt = tTok + TILE_M;
  float*  tW   = reinterpret_cast<float*>(tSlt + TILE_M);

  // --- tile metadata (pad partial tiles by replicating last valid row) ---
  if (threadIdx.x < TILE_M) {
    int m = threadIdx.x;
    int r = ts + (m < tc - 1 ? m : tc - 1);
    tTok[m] = rowTok[r];
    tSlt[m] = rowSlot[r];
    tW[m]   = rowW[r];
  }
  __syncthreads();

  // --- gather X tile: 8 threads per row, interleaved float4, cvt -> bf16 ---
  {
    int mrow = threadIdx.x >> 3;
    int j    = threadIdx.x & 7;
    const float4* xs = reinterpret_cast<const float4*>(x + (size_t)tTok[mrow] * D_MODEL);
    __bf16* ax = Ax + mrow * AXS;
#pragma unroll 4
    for (int i = 0; i < 32; ++i) {
      float4 v = xs[j + i * 8];
      int k = (j + i * 8) * 4;
      ax[k + 0] = (__bf16)v.x;
      ax[k + 1] = (__bf16)v.y;
      ax[k + 2] = (__bf16)v.z;
      ax[k + 3] = (__bf16)v.w;
    }
  }
  __syncthreads();

  int lane = threadIdx.x & 31;
  int wv   = threadIdx.x >> 5;        // 0..15
  int hl   = lane & 15;
  int hi8  = (lane >> 4) * 8;
  int hi16 = (lane >> 4) * 16;

  v8f acc2[16];                       // 64 x (16 waves * 64 cols) -> 128 VGPRs
#pragma unroll
  for (int i = 0; i < 16; ++i) acc2[i] = zerov8();

#pragma unroll 1
  for (int fc = 0; fc < D_FF / FCHUNK; ++fc) {
    int fbase = fc * FCHUNK;

    // ---- GEMM1: this wave owns 16 F-columns [fbase + wv*16, +16) ----
    v8f hacc[4];
#pragma unroll
    for (int i = 0; i < 4; ++i) hacc[i] = zerov8();

    const __bf16* bp = W1t + ((size_t)e * D_FF + (fbase + wv * 16 + hl)) * D_MODEL;

    // Pipelines: B fragment 1 k-step ahead (global), A fragments 2 WMMA-steps
    // ahead through a 4-deep register ring (ds), refills placed AFTER the
    // consuming WMMA so no WMMA->ds WAR hazard nops are needed.
    v16bf bcur = ldfragB(bp, 0, hi16);
    v16bf abuf[4];
    abuf[0] = ldfragA(Ax + (0 * 16 + hl) * AXS, 0, hi8);
    abuf[1] = ldfragA(Ax + (1 * 16 + hl) * AXS, 0, hi8);

#pragma unroll 2
    for (int k0 = 0; k0 < D_MODEL; k0 += 32) {
      int kn = (k0 + 32 < D_MODEL) ? (k0 + 32) : k0;
      __builtin_prefetch(bp + k0 + 512, 0, 1);
      v16bf bnxt = ldfragB(bp, kn, hi16);
#pragma unroll
      for (int mt = 0; mt < 4; ++mt) {
        hacc[mt] = WMMA_BF16(abuf[mt], bcur, hacc[mt]);
        int mt2 = (mt + 2) & 3;                // refill 2 steps ahead
        int ka  = (mt < 2) ? k0 : kn;          // mt2 belongs to this/next k-step
        abuf[mt2] = ldfragA(Ax + (mt2 * 16 + hl) * AXS, ka, hi8);
      }
      bcur = bnxt;
    }

    float bias1 = b1[e * D_FF + fbase + wv * 16 + hl];
    __syncthreads();   // previous chunk's GEMM2 done reading Ac
#pragma unroll
    for (int mt = 0; mt < 4; ++mt) {
#pragma unroll
      for (int r = 0; r < 8; ++r) {
        float h = hacc[mt][r] + bias1;
        float g = 0.5f * h * (1.0f + erff(h * 0.70710678118654752f));  // exact GELU
        Ac[(mt * 16 + r + hi8) * ACS + wv * 16 + hl] = (__bf16)g;
      }
    }
    __syncthreads();

    // ---- GEMM2 partial: this wave owns 64 d-columns [wv*64, +64) ----
    // A fragments ping-pong between two 4-frag sets; the refill for k-step
    // ks+1 issues after the nt=2/3 WMMA groups of k-step ks (>=4 WMMAs of
    // hazard distance, >=2 WMMAs of load latency cover). B fragment is
    // pipelined 1 (k,nt)-step ahead as before.
    const __bf16* b2base =
        W2t + ((size_t)e * D_MODEL + (wv * 64 + hl)) * D_FF + fbase;
    v16bf afA[4], afB[4];
#pragma unroll
    for (int mt = 0; mt < 4; ++mt)
      afA[mt] = ldfragA(Ac + (mt * 16 + hl) * ACS, 0, hi8);
    v16bf b2cur = ldfragB(b2base, 0, hi16);

#pragma unroll 1
    for (int ks2 = 0; ks2 < 4; ++ks2) {
#pragma unroll
      for (int half = 0; half < 2; ++half) {
        int ks = ks2 * 2 + half;
        int k0 = ks * 32;
        int kn = (k0 + 32 < FCHUNK) ? (k0 + 32) : k0;   // dummy reload at end
        v16bf (&cur)[4] = half ? afB : afA;
        v16bf (&nxt)[4] = half ? afA : afB;
#pragma unroll
        for (int nt = 0; nt < 4; ++nt) {
          int it  = ks * 4 + nt;
          int itn = (it + 1 < 32) ? (it + 1) : it;
          int knB = (itn >> 2) * 32;
          int ntn = itn & 3;
          v16bf b2n = ldfragB(b2base + (size_t)ntn * 16 * D_FF, knB, hi16);
#pragma unroll
          for (int mt = 0; mt < 4; ++mt)
            acc2[mt * 4 + nt] = WMMA_BF16(cur[mt], b2cur, acc2[mt * 4 + nt]);
          if (nt == 2) {
            nxt[0] = ldfragA(Ac + (0 * 16 + hl) * ACS, kn, hi8);
            nxt[1] = ldfragA(Ac + (1 * 16 + hl) * ACS, kn, hi8);
          }
          if (nt == 3) {
            nxt[2] = ldfragA(Ac + (2 * 16 + hl) * ACS, kn, hi8);
            nxt[3] = ldfragA(Ac + (3 * 16 + hl) * ACS, kn, hi8);
          }
          b2cur = b2n;
        }
      }
    }
  }

  // ---- epilogue: bias2, gate weight, scatter to ybuf[(t*2+slot)] ----
#pragma unroll
  for (int nt = 0; nt < 4; ++nt) {
    int d = wv * 64 + nt * 16 + hl;
    float bias2 = b2[e * D_MODEL + d];
#pragma unroll
    for (int mt = 0; mt < 4; ++mt) {
#pragma unroll
      for (int r = 0; r < 8; ++r) {
        int m = mt * 16 + r + hi8;
        if (m < tc) {
          int tok = tTok[m];
          int sl  = tSlt[m];
          float wg = tW[m];
          ybuf[((size_t)tok * 2 + sl) * D_MODEL + d] = wg * (acc2[mt * 4 + nt][r] + bias2);
        }
      }
    }
  }
}

// ---------------------------------------------------------------------------
// K6: combine the two routed contributions per token (fixed order -> determ.)
// ---------------------------------------------------------------------------
__global__ __launch_bounds__(256) void combine_kernel(const float* __restrict__ ybuf,
                                                      float* __restrict__ out)
{
  int i = blockIdx.x * 256 + threadIdx.x;       // float4 index over T*D/4
  int t = i >> 8;                               // D/4 = 256 float4 per token
  int c = i & 255;
  const float4* y = reinterpret_cast<const float4*>(ybuf);
  float4 a = y[((size_t)t * 2 + 0) * 256 + c];
  float4 b = y[((size_t)t * 2 + 1) * 256 + c];
  float4 o;
  o.x = a.x + b.x; o.y = a.y + b.y; o.z = a.z + b.z; o.w = a.w + b.w;
  reinterpret_cast<float4*>(out)[i] = o;
}

// ---------------------------------------------------------------------------
// Workspace layout (bytes). Total ~193 MB.
// ---------------------------------------------------------------------------
#define OFF_COUNTS   0          // 8 int
#define OFF_FILL     64         // 8 int
#define OFF_OFFSETS  128        // 9 int
#define OFF_NUMTILES 256        // 1 int
#define OFF_TILE_E   512        // MAX_TILES int
#define OFF_TILE_S   2048
#define OFF_TILE_C   4096
#define OFF_TOK_E    8192                        // T*2 int
#define OFF_TOK_W    (OFF_TOK_E   + T_TOK*2*4)
#define OFF_ROW_TOK  (OFF_TOK_W   + T_TOK*2*4)
#define OFF_ROW_SLOT (OFF_ROW_TOK + T_TOK*2*4)
#define OFF_ROW_W    (OFF_ROW_SLOT+ T_TOK*2*4)
#define OFF_W1T      335872                      // aligned; 8*4096*1024*2 bytes
#define OFF_W2T      (OFF_W1T + (size_t)N_EXP*D_FF*D_MODEL*2)
#define OFF_YBUF     (OFF_W2T + (size_t)N_EXP*D_MODEL*D_FF*2)

extern "C" void kernel_launch(void* const* d_in, const int* in_sizes, int n_in,
                              void* d_out, int out_size, void* d_ws, size_t ws_size,
                              hipStream_t stream) {
  (void)in_sizes; (void)n_in; (void)out_size; (void)ws_size;

  const float* x  = (const float*)d_in[0];
  const float* Wg = (const float*)d_in[1];
  const float* bg = (const float*)d_in[2];
  const float* W1 = (const float*)d_in[3];
  const float* b1 = (const float*)d_in[4];
  const float* W2 = (const float*)d_in[5];
  const float* b2 = (const float*)d_in[6];

  float* out      = (float*)d_out;
  float* gate_out = out + (size_t)T_TOK * D_MODEL;   // outputs concatenated: (out, gate_logits)

  char* ws = (char*)d_ws;
  int*    counts   = (int*)(ws + OFF_COUNTS);
  int*    fill     = (int*)(ws + OFF_FILL);
  int*    offsets  = (int*)(ws + OFF_OFFSETS);
  int*    numTiles = (int*)(ws + OFF_NUMTILES);
  int*    tileE    = (int*)(ws + OFF_TILE_E);
  int*    tileS    = (int*)(ws + OFF_TILE_S);
  int*    tileC    = (int*)(ws + OFF_TILE_C);
  int*    tokE     = (int*)(ws + OFF_TOK_E);
  float*  tokW     = (float*)(ws + OFF_TOK_W);
  int*    rowTok   = (int*)(ws + OFF_ROW_TOK);
  int*    rowSlot  = (int*)(ws + OFF_ROW_SLOT);
  float*  rowW     = (float*)(ws + OFF_ROW_W);
  __bf16* W1t      = (__bf16*)(ws + OFF_W1T);
  __bf16* W2t      = (__bf16*)(ws + OFF_W2T);
  float*  ybuf     = (float*)(ws + OFF_YBUF);

  (void)hipMemsetAsync(d_ws, 0, 8192, stream);   // counts/fill/offsets/numTiles

  // one-time (per call) weight convert+transpose: bf16 with B-columns contiguous
  transpose_cvt_kernel<<<dim3(D_FF / 32, D_MODEL / 32, N_EXP), dim3(32, 8), 0, stream>>>(
      W1, W1t, D_MODEL, D_FF);
  transpose_cvt_kernel<<<dim3(D_MODEL / 32, D_FF / 32, N_EXP), dim3(32, 8), 0, stream>>>(
      W2, W2t, D_FF, D_MODEL);

  gate_kernel<<<T_TOK / 8, 256, 0, stream>>>(x, Wg, bg, gate_out, counts, tokE, tokW);
  plan_kernel<<<1, 64, 0, stream>>>(counts, offsets, tileE, tileS, tileC, numTiles);
  scatter_kernel<<<T_TOK / 256, 256, 0, stream>>>(tokE, tokW, offsets, fill,
                                                  rowTok, rowSlot, rowW);

  (void)hipFuncSetAttribute(reinterpret_cast<const void*>(moe_expert_kernel),
                            hipFuncAttributeMaxDynamicSharedMemorySize, SMEM_BYTES);
  moe_expert_kernel<<<MAX_TILES, 512, SMEM_BYTES, stream>>>(
      x, W1t, W2t, b1, b2, numTiles, tileE, tileS, tileC, rowTok, rowSlot, rowW, ybuf);

  combine_kernel<<<(T_TOK * D_MODEL / 4) / 256, 256, 0, stream>>>(ybuf, out);
}
